// SPVCNN_40759239639905
// MI455X (gfx1250) — matrix operand order repaired and checked by reading, not compiled, not run on hardware
//
#include <hip/hip_runtime.h>

// ---------------------------------------------------------------------------
// SPVCNN forward for gfx1250 (MI455X).  Implicit-GEMM convs through
// V_WMMA_F32_16X16X32_BF16 (wave32).  Activations live in bf16 end-to-end so
// the conv inner loop is pure b128 loads + WMMA (no conversion VALU); conv
// outputs accumulate/store fp32 into one scratch (= d_out), and masked-BN
// statistics are computed in fp32 from it.  Weights are prepacked once to
// bf16 in B-fragment register order.  Each wave: 16 voxels x 32 couts.
// ---------------------------------------------------------------------------

typedef __attribute__((ext_vector_type(16))) __bf16 v16bf;
typedef __attribute__((ext_vector_type(8)))  float  v8f;

#define BN_EPS 1e-5f

__device__ __forceinline__ unsigned short f2bf(float f) {
  union { float f; unsigned u; } x; x.f = f;
  unsigned r = x.u + 0x7FFFu + ((x.u >> 16) & 1u);   // round-to-nearest-even
  return (unsigned short)(r >> 16);
}
__device__ __forceinline__ float bf2f(unsigned short u) {
  union { unsigned u; float f; } x; x.u = ((unsigned)u) << 16;
  return x.f;
}

union Frag {
  v16bf v;
  unsigned short u[16];
  uint4 q[2];
};

// ---------------------------------------------------------------------------
// Weight prepack: [k,k,k,Cin,Cout] fp32 -> bf16 tiles in B-fragment order.
//   value(lane, e) = w[tap, kc*32 + (lane>>4)*8 + (e&7) + (e>>3)*16, ct*16 + (lane&15)]
// zero-padded where cin >= Cin (stem: Cin=4 padded to 32).
// ---------------------------------------------------------------------------
__global__ __launch_bounds__(256)
void prepack_w(const float* __restrict__ w, unsigned short* __restrict__ pw,
               int taps, int Cin, int Cout)
{
  const int kchunks = (Cin + 31) >> 5;
  const int nct = Cout >> 4;
  const long long total = (long long)taps * kchunks * nct * 512;
  for (long long i = (long long)blockIdx.x * blockDim.x + threadIdx.x; i < total;
       i += (long long)gridDim.x * blockDim.x) {
    const int e    = (int)(i & 15);
    const int lane = (int)((i >> 4) & 31);
    long long t = i >> 9;
    const int ct  = (int)(t % nct);  t /= nct;
    const int kc  = (int)(t % kchunks);
    const int tap = (int)(t / kchunks);
    const int cin  = kc * 32 + ((lane >> 4) << 3) + (e & 7) + ((e >> 3) << 4);
    const int cout = ct * 16 + (lane & 15);
    pw[i] = (cin < Cin) ? f2bf(w[((size_t)tap * Cin + cin) * Cout + cout])
                        : (unsigned short)0;
  }
}

// ---------------------------------------------------------------------------
// Implicit-GEMM conv.  KT=0: 3x3x3 SAME; KT=1: 2x2x2 pad(0,1); KT=2: 1x1x1.
// Input bf16 [D,D,D,Cin], Cin%32==0; output fp32 [D,D,D,Cout], Cout%32==0.
// Block = 256 = 8 waves; wave owns 16 voxels x 2 cout tiles.  Out-of-bounds
// taps read a zeroed bf16 row (branchless pointer select).
// Inner loop: 2 b128 (A) + 4 b128 (B) + 2 WMMA, no conversion ops.
// ---------------------------------------------------------------------------
template<int KT>
__global__ __launch_bounds__(256)
void conv3d_wmma(const unsigned short* __restrict__ in,
                 const unsigned short* __restrict__ pw,
                 const unsigned short* __restrict__ zrow,
                 float* __restrict__ out, int D, int Cin, int Cout)
{
  constexpr int KK   = (KT == 0) ? 3 : (KT == 1 ? 2 : 1);
  constexpr int TAPS = KK * KK * KK;

  const int lane    = threadIdx.x & 31;
  const int wave    = threadIdx.x >> 5;
  const int voxBase = (blockIdx.x * 8 + wave) * 16;
  const int nct     = Cout >> 4;
  const int ct0     = blockIdx.y * 2;
  const int kchunks = Cin >> 5;

  // A-matrix 16x32 bf16 layout: both half-waves hold rows 0..15;
  // lanes 0-15 carry K 0-7 & 16-23, lanes 16-31 carry K 8-15 & 24-31.
  const int M   = lane & 15;
  const int vox = voxBase + M;
  const int X = vox % D;
  const int Y = (vox / D) % D;
  const int Z = vox / (D * D);
  const int kbase = (lane >> 4) * 8;
  const int ncol  = lane & 15;

  v8f acc0 = {}, acc1 = {};

  for (int tap = 0; tap < TAPS; ++tap) {
    int dz, dy, dx;
    if (KT == 0)      { dz = tap / 9 - 1; dy = (tap / 3) % 3 - 1; dx = tap % 3 - 1; }
    else if (KT == 1) { dz = (tap >> 2) & 1; dy = (tap >> 1) & 1; dx = tap & 1; }
    else              { dz = 0; dy = 0; dx = 0; }
    const int nz = Z + dz, ny = Y + dy, nx = X + dx;
    const bool valid = (unsigned)nz < (unsigned)D && (unsigned)ny < (unsigned)D &&
                       (unsigned)nx < (unsigned)D;
    const unsigned short* src = valid
        ? (in + ((size_t)(nz * D + ny) * D + nx) * Cin + kbase)
        : (zrow + kbase);
    __builtin_prefetch(src, 0, 3);   // global_prefetch into near caches

    const unsigned short* bp =
        pw + ((size_t)tap * kchunks * nct + ct0) * 512 + lane * 16;

    for (int kc = 0; kc < kchunks; ++kc) {
      // ---- A fragment: two 16B runs of bf16, no conversion ----
      Frag a;
      a.q[0] = *(const uint4*)(src + kc * 32);
      a.q[1] = *(const uint4*)(src + kc * 32 + 16);

      // ---- B fragments: prepacked, 32B contiguous per lane ----
      const unsigned short* bpk = bp + (size_t)kc * nct * 512;
      Frag b0, b1;
      b0.q[0] = *(const uint4*)(bpk);
      b0.q[1] = *(const uint4*)(bpk + 8);
      b1.q[0] = *(const uint4*)(bpk + 512);
      b1.q[1] = *(const uint4*)(bpk + 520);

      acc0 = __builtin_amdgcn_wmma_f32_16x16x32_bf16(false, a.v, false, b0.v,
                                                     (short)0, acc0, false, false);
      acc1 = __builtin_amdgcn_wmma_f32_16x16x32_bf16(false, a.v, false, b1.v,
                                                     (short)0, acc1, false, false);
    }
  }

  // D layout: VGPR r holds rows r / r+8 per half-wave, col = lane&15
  const int mrow = (lane >> 4) << 3;
  float* o = out + (size_t)(voxBase + mrow) * Cout + ct0 * 16 + ncol;
  #pragma unroll
  for (int r = 0; r < 8; ++r) {
    o[(size_t)r * Cout]      = acc0[r];
    o[(size_t)r * Cout + 16] = acc1[r];
  }
}

// ---------------------------------------------------------------------------
// Masked batch-norm statistics over fp32 conv output.
// ---------------------------------------------------------------------------
__global__ __launch_bounds__(256)
void bn_stats(const float* __restrict__ x, const float* __restrict__ m,
              float* __restrict__ stats, int N, int C, int slices)
{
  const int c = blockIdx.x / slices;
  const int s = blockIdx.x % slices;
  float sum = 0.f, sq = 0.f;
  for (long long v = (long long)s * blockDim.x + threadIdx.x; v < N;
       v += (long long)blockDim.x * slices) {
    const float xv = x[v * (long long)C + c] * m[v];
    sum += xv; sq += xv * xv;
  }
  for (int off = 16; off; off >>= 1) {
    sum += __shfl_xor(sum, off, 32);
    sq  += __shfl_xor(sq,  off, 32);
  }
  if ((threadIdx.x & 31) == 0) {
    atomicAdd(&stats[c], sum);
    atomicAdd(&stats[C + c], sq);
  }
}

__global__ __launch_bounds__(256)
void count_mask(const float* __restrict__ m, float* __restrict__ cnt, int N)
{
  float s = 0.f;
  for (long long v = (long long)blockIdx.x * blockDim.x + threadIdx.x; v < N;
       v += (long long)gridDim.x * blockDim.x)
    s += m[v];
  for (int off = 16; off; off >>= 1) s += __shfl_xor(s, off, 32);
  if ((threadIdx.x & 31) == 0) atomicAdd(cnt, s);
}

// y = ((x - mean) * rsqrt(var+eps) * g + b) * m, optional ReLU; bf16 out.
__global__ __launch_bounds__(256)
void bn_apply(const float* __restrict__ x, const float* __restrict__ m,
              const float* __restrict__ stats, const float* __restrict__ cnt,
              const float* __restrict__ gamma, const float* __restrict__ beta,
              unsigned short* __restrict__ out, long long N, int C, int relu)
{
  const float cn = fmaxf(cnt[0], 1.f);
  const long long total = N * C;
  for (long long i = (long long)blockIdx.x * blockDim.x + threadIdx.x; i < total;
       i += (long long)gridDim.x * blockDim.x) {
    const int c = (int)(i % C);
    const long long v = i / C;
    const float mean = stats[c] / cn;
    const float var  = stats[C + c] / cn - mean * mean;
    float y = (x[i] - mean) * rsqrtf(var + BN_EPS) * gamma[c] + beta[c];
    y *= m[v];
    if (relu) y = fmaxf(y, 0.f);
    out[i] = f2bf(y);
  }
}

__global__ __launch_bounds__(256)
void add_relu_bf(const unsigned short* __restrict__ a,
                 const unsigned short* __restrict__ b,
                 unsigned short* __restrict__ out, long long total)
{
  for (long long i = (long long)blockIdx.x * blockDim.x + threadIdx.x; i < total;
       i += (long long)gridDim.x * blockDim.x)
    out[i] = f2bf(fmaxf(bf2f(a[i]) + bf2f(b[i]), 0.f));
}

// final: fp32 output into d_out
__global__ __launch_bounds__(256)
void add_relu_f32(const unsigned short* __restrict__ a,
                  const unsigned short* __restrict__ b,
                  float* __restrict__ out, long long total)
{
  for (long long i = (long long)blockIdx.x * blockDim.x + threadIdx.x; i < total;
       i += (long long)gridDim.x * blockDim.x)
    out[i] = fmaxf(bf2f(a[i]) + bf2f(b[i]), 0.f);
}

__global__ __launch_bounds__(256)
void mask_to_float(const int* __restrict__ mi, float* __restrict__ mf, long long n)
{
  for (long long i = (long long)blockIdx.x * blockDim.x + threadIdx.x; i < n;
       i += (long long)gridDim.x * blockDim.x)
    mf[i] = (float)mi[i];
}

// xin[v, 0:32] = { bf16(feats[v,c]*m[v]) for c<4, else 0 }
__global__ __launch_bounds__(256)
void mask_mul_pad(const float* __restrict__ x, const float* __restrict__ m,
                  unsigned short* __restrict__ out, long long N, int Cin, int Cpad)
{
  const long long total = N * Cpad;
  for (long long i = (long long)blockIdx.x * blockDim.x + threadIdx.x; i < total;
       i += (long long)gridDim.x * blockDim.x) {
    const int c = (int)(i % Cpad);
    const long long v = i / Cpad;
    out[i] = (c < Cin) ? f2bf(x[v * Cin + c] * m[v]) : (unsigned short)0;
  }
}

// sum-pool 2x2x2 (bf16 feats, fp32 accumulate)
__global__ __launch_bounds__(256)
void pool_feats(const unsigned short* __restrict__ in,
                unsigned short* __restrict__ out, int Do, int C)
{
  const int Di = Do * 2;
  const long long total = (long long)Do * Do * Do * C;
  for (long long i = (long long)blockIdx.x * blockDim.x + threadIdx.x; i < total;
       i += (long long)gridDim.x * blockDim.x) {
    const int c = (int)(i % C);
    long long vo = i / C;
    const int xo = (int)(vo % Do), yo = (int)((vo / Do) % Do), zo = (int)(vo / ((long long)Do * Do));
    float s = 0.f;
    #pragma unroll
    for (int d = 0; d < 8; ++d) {
      const int dz = (d >> 2) & 1, dy = (d >> 1) & 1, dx = d & 1;
      s += bf2f(in[((size_t)((zo * 2 + dz) * Di + yo * 2 + dy) * Di + xo * 2 + dx) * C + c]);
    }
    out[i] = f2bf(s);
  }
}

// max-pool 2x2x2 (fp32 mask)
__global__ __launch_bounds__(256)
void pool_mask(const float* __restrict__ in, float* __restrict__ out, int Do)
{
  const int Di = Do * 2;
  const long long total = (long long)Do * Do * Do;
  for (long long vo = (long long)blockIdx.x * blockDim.x + threadIdx.x; vo < total;
       vo += (long long)gridDim.x * blockDim.x) {
    const int xo = (int)(vo % Do), yo = (int)((vo / Do) % Do), zo = (int)(vo / ((long long)Do * Do));
    float s = 0.f;
    #pragma unroll
    for (int d = 0; d < 8; ++d) {
      const int dz = (d >> 2) & 1, dy = (d >> 1) & 1, dx = d & 1;
      s = fmaxf(s, in[(size_t)((zo * 2 + dz) * Di + yo * 2 + dy) * Di + xo * 2 + dx]);
    }
    out[vo] = s;
  }
}

// nearest-neighbor 2x upsample (bf16 feats)
__global__ __launch_bounds__(256)
void upsample_feats(const unsigned short* __restrict__ in,
                    unsigned short* __restrict__ out, int Do, int C)
{
  const int Di = Do / 2;
  const long long total = (long long)Do * Do * Do * C;
  for (long long i = (long long)blockIdx.x * blockDim.x + threadIdx.x; i < total;
       i += (long long)gridDim.x * blockDim.x) {
    const int c = (int)(i % C);
    long long v = i / C;
    const int x = (int)(v % Do), y = (int)((v / Do) % Do), z = (int)(v / ((long long)Do * Do));
    out[i] = in[((size_t)((z >> 1) * Di + (y >> 1)) * Di + (x >> 1)) * C + c];
  }
}

// nearest-neighbor 2x upsample (fp32 mask)
__global__ __launch_bounds__(256)
void upsample_mask(const float* __restrict__ in, float* __restrict__ out, int Do)
{
  const int Di = Do / 2;
  const long long total = (long long)Do * Do * Do;
  for (long long v = (long long)blockIdx.x * blockDim.x + threadIdx.x; v < total;
       v += (long long)gridDim.x * blockDim.x) {
    const int x = (int)(v % Do), y = (int)((v / Do) % Do), z = (int)(v / ((long long)Do * Do));
    out[v] = in[(size_t)((z >> 1) * Di + (y >> 1)) * Di + (x >> 1)];
  }
}

// out[v, 0:Ca] = a[v];  out[v, Ca:Ca+Cb] = b[v] * m[v]   (bf16)
__global__ __launch_bounds__(256)
void concat_masked(const unsigned short* __restrict__ a, int Ca,
                   const unsigned short* __restrict__ b, int Cb,
                   const float* __restrict__ m, unsigned short* __restrict__ out,
                   long long N)
{
  const int C = Ca + Cb;
  const long long total = N * C;
  for (long long i = (long long)blockIdx.x * blockDim.x + threadIdx.x; i < total;
       i += (long long)gridDim.x * blockDim.x) {
    const long long v = i / C;
    const int c = (int)(i % C);
    out[i] = (c < Ca) ? a[v * Ca + c]
                      : f2bf(bf2f(b[v * Cb + (c - Ca)]) * m[v]);
  }
}

// ---------------------------------------------------------------------------
// Host orchestration
// ---------------------------------------------------------------------------
static inline int ew_blocks(long long total) {
  long long b = (total + 255) / 256;
  if (b > 8192) b = 8192;
  if (b < 1) b = 1;
  return (int)b;
}

extern "C" void kernel_launch(void* const* d_in, const int* in_sizes, int n_in,
                              void* d_out, int out_size, void* d_ws, size_t ws_size,
                              hipStream_t stream)
{
  (void)in_sizes; (void)n_in; (void)out_size; (void)ws_size;

  const float* feats = (const float*)d_in[0];
  const int*   maski = (const int*)d_in[1];
  auto W = [&](int i) { return (const float*)d_in[i]; };

  const int D0 = 64, D1 = 32, D2 = 16;
  const long long N0 = (long long)D0 * D0 * D0;
  const long long N1 = (long long)D1 * D1 * D1;
  const long long N2 = (long long)D2 * D2 * D2;
  const int C0 = 32, C1 = 64, C2 = 128, C3 = 96, C4 = 96, CIN = 4;

  // --- workspace bump allocator (~220 MB) ---
  float* wsf = (float*)d_ws;
  size_t off = 0;
  auto allocF = [&](size_t n) { float* p = wsf + off; off += (n + 3) & ~(size_t)3; return p; };
  auto allocU = [&](size_t n) { return (unsigned short*)allocF((n + 1) / 2); };

  float* m0f   = allocF(N0);
  float* m1f   = allocF(N1);
  float* m2f   = allocF(N2);
  float* mu1f  = allocF(N1);
  float* mu2f  = allocF(N0);
  float* cnts  = allocF(8);
  float* stats = allocF(256);
  unsigned short* zrow = allocU(256);              // zeroed boundary row
  unsigned short* xin  = allocU((size_t)N0 * 32);  // stem input padded 4 -> 32 ch
  unsigned short* x0   = allocU((size_t)N0 * C0);  // skip for up2
  unsigned short* x1   = allocU((size_t)N1 * C1);  // skip for up1
  unsigned short* actA = allocU((size_t)N0 * 96);
  unsigned short* actB = allocU((size_t)N0 * 96);  // main bf16 stream
  unsigned short* actCat = allocU((size_t)N0 * 128);
  float* TC = (float*)d_out;                       // fp32 conv scratch == d_out

  hipMemsetAsync(zrow, 0, 256 * sizeof(unsigned short), stream);

  // --- prepack all conv weights to bf16 fragment order ---
  struct LayerW { int idx, taps, cin, cout; };
  static const LayerW LW[] = {
    {2,27,4,32},   {5,8,32,32},
    {8,27,32,64},  {9,27,64,64},   {14,1,32,64},
    {17,27,64,64}, {18,27,64,64},
    {23,8,64,64},
    {26,27,64,128},{27,27,128,128},{32,1,64,128},
    {35,27,128,128},{36,27,128,128},
    {41,27,128,96},
    {44,27,160,96},{45,27,96,96},  {50,1,160,96},
    {53,27,96,96}, {54,27,96,96},
    {59,27,96,96},
    {62,27,128,96},{63,27,96,96},  {68,1,128,96},
    {71,27,96,96}, {72,27,96,96},
  };
  unsigned short* pw[80] = {};
  for (int l = 0; l < (int)(sizeof(LW) / sizeof(LW[0])); ++l) {
    const LayerW& L = LW[l];
    const int kch = (L.cin + 31) >> 5;
    const long long nus = (long long)L.taps * kch * (L.cout >> 4) * 512;
    unsigned short* p = allocU((size_t)nus);
    pw[L.idx] = p;
    prepack_w<<<dim3(ew_blocks(nus)), dim3(256), 0, stream>>>(
        W(L.idx), p, L.taps, L.cin, L.cout);
  }

  // --- launch helpers ---
  auto conv = [&](int kt, const unsigned short* in, int widx, float* out,
                  int D, int CinPad, int Cout) {
    dim3 g((unsigned)((long long)D * D * D / 128), (unsigned)(Cout / 32));
    dim3 b(256);
    if (kt == 0)      conv3d_wmma<0><<<g, b, 0, stream>>>(in, pw[widx], zrow, out, D, CinPad, Cout);
    else if (kt == 1) conv3d_wmma<1><<<g, b, 0, stream>>>(in, pw[widx], zrow, out, D, CinPad, Cout);
    else              conv3d_wmma<2><<<g, b, 0, stream>>>(in, pw[widx], zrow, out, D, CinPad, Cout);
  };
  auto bn = [&](const float* x, const float* m, const float* g, const float* b,
                const float* cnt, long long N, int C, bool relu, unsigned short* out) {
    hipMemsetAsync(stats, 0, 2 * C * sizeof(float), stream);
    const int slices = (N >= 262144) ? 32 : (N >= 32768 ? 8 : 2);
    bn_stats<<<dim3((unsigned)(C * slices)), dim3(256), 0, stream>>>(
        x, m, stats, (int)N, C, slices);
    bn_apply<<<dim3(ew_blocks(N * C)), dim3(256), 0, stream>>>(
        x, m, stats, cnt, g, b, out, N, C, relu ? 1 : 0);
  };
  auto count = [&](const float* m, float* cnt, long long N) {
    hipMemsetAsync(cnt, 0, sizeof(float), stream);
    count_mask<<<dim3(ew_blocks(N)), dim3(256), 0, stream>>>(m, cnt, (int)N);
  };
  // residual block: inb -> outb (bf16).  h in actA, shortcut in actCat.
  auto resblock = [&](unsigned short* inb, int D, long long N, int Cin, int Cout,
                      int w1, const float* g1, const float* b1,
                      int w2, const float* g2, const float* b2,
                      int wd, const float* gd, const float* bd,
                      const float* m, const float* cnt, unsigned short* outb) {
    conv(0, inb, w1, TC, D, Cin, Cout);
    bn(TC, m, g1, b1, cnt, N, Cout, true, actA);                   // h
    if (wd >= 0) {
      conv(2, inb, wd, TC, D, Cin, Cout);                          // shortcut
      bn(TC, m, gd, bd, cnt, N, Cout, false, actCat);
    }
    conv(0, actA, w2, TC, D, Cout, Cout);
    bn(TC, m, g2, b2, cnt, N, Cout, false, actA);                  // h2 (in place)
    const unsigned short* d = (wd >= 0) ? actCat : inb;
    add_relu_bf<<<dim3(ew_blocks(N * Cout)), dim3(256), 0, stream>>>(
        actA, d, outb, N * Cout);
  };

  // --- masks & counts ---
  mask_to_float<<<dim3(ew_blocks(N0)), dim3(256), 0, stream>>>(maski, m0f, N0);
  count(m0f, &cnts[0], N0);

  // --- stem ---
  mask_mul_pad<<<dim3(ew_blocks(N0 * 32)), dim3(256), 0, stream>>>(
      feats, m0f, xin, N0, CIN, 32);
  conv(0, xin, 2, TC, D0, 32, C0);
  bn(TC, m0f, W(3), W(4), &cnts[0], N0, C0, true, x0);

  // --- stage1 (stride 2) ---
  pool_feats<<<dim3(ew_blocks(N1 * C0)), dim3(256), 0, stream>>>(x0, actA, D1, C0);
  pool_mask<<<dim3(ew_blocks(N1)), dim3(256), 0, stream>>>(m0f, m1f, D1);
  count(m1f, &cnts[1], N1);
  conv(1, actA, 5, TC, D1, C0, C0);
  bn(TC, m1f, W(6), W(7), &cnts[1], N1, C0, true, actB);
  resblock(actB, D1, N1, C0, C1, 8, W(10), W(11), 9, W(12), W(13),
           14, W(15), W(16), m1f, &cnts[1], actB);                 // s1_rb1
  resblock(actB, D1, N1, C1, C1, 17, W(19), W(20), 18, W(21), W(22),
           -1, nullptr, nullptr, m1f, &cnts[1], x1);               // s1_rb2 -> x1

  // --- stage2 (stride 4) ---
  pool_feats<<<dim3(ew_blocks(N2 * C1)), dim3(256), 0, stream>>>(x1, actA, D2, C1);
  pool_mask<<<dim3(ew_blocks(N2)), dim3(256), 0, stream>>>(m1f, m2f, D2);
  count(m2f, &cnts[2], N2);
  conv(1, actA, 23, TC, D2, C1, C1);
  bn(TC, m2f, W(24), W(25), &cnts[2], N2, C1, true, actB);
  resblock(actB, D2, N2, C1, C2, 26, W(28), W(29), 27, W(30), W(31),
           32, W(33), W(34), m2f, &cnts[2], actB);                 // s2_rb1
  resblock(actB, D2, N2, C2, C2, 35, W(37), W(38), 36, W(39), W(40),
           -1, nullptr, nullptr, m2f, &cnts[2], actB);             // s2_rb2 -> x2

  // --- up1: deconv to stride 2 + fuse with x1 ---
  upsample_mask<<<dim3(ew_blocks(N1)), dim3(256), 0, stream>>>(m2f, mu1f, D1);
  count(mu1f, &cnts[3], N1);
  upsample_feats<<<dim3(ew_blocks(N1 * C2)), dim3(256), 0, stream>>>(actB, actA, D1, C2);
  conv(0, actA, 41, TC, D1, C2, C3);
  bn(TC, mu1f, W(42), W(43), &cnts[3], N1, C3, true, actB);        // y3a
  concat_masked<<<dim3(ew_blocks(N1 * (C3 + C1))), dim3(256), 0, stream>>>(
      actB, C3, x1, C1, mu1f, actCat, N1);
  {
    // u1_rb1 on concat (160 -> 96); shortcut parks in actCat after it's read
    conv(0, actCat, 44, TC, D1, C3 + C1, C3);
    bn(TC, mu1f, W(46), W(47), &cnts[3], N1, C3, true, actA);
    conv(2, actCat, 50, TC, D1, C3 + C1, C3);
    bn(TC, mu1f, W(51), W(52), &cnts[3], N1, C3, false, actCat);
    conv(0, actA, 45, TC, D1, C3, C3);
    bn(TC, mu1f, W(48), W(49), &cnts[3], N1, C3, false, actA);
    add_relu_bf<<<dim3(ew_blocks(N1 * C3)), dim3(256), 0, stream>>>(
        actA, actCat, actB, N1 * C3);
  }
  resblock(actB, D1, N1, C3, C3, 53, W(55), W(56), 54, W(57), W(58),
           -1, nullptr, nullptr, mu1f, &cnts[3], actB);            // u1_rb2 -> y3

  // --- up2: deconv to stride 1 + fuse with x0 ---
  upsample_mask<<<dim3(ew_blocks(N0)), dim3(256), 0, stream>>>(mu1f, mu2f, D0);
  count(mu2f, &cnts[4], N0);
  upsample_feats<<<dim3(ew_blocks(N0 * C3)), dim3(256), 0, stream>>>(actB, actA, D0, C3);
  conv(0, actA, 59, TC, D0, C3, C4);
  bn(TC, mu2f, W(60), W(61), &cnts[4], N0, C4, true, actB);        // y4a
  concat_masked<<<dim3(ew_blocks(N0 * (C4 + C0))), dim3(256), 0, stream>>>(
      actB, C4, x0, C0, mu2f, actCat, N0);
  {
    // u2_rb1 on concat (128 -> 96)
    conv(0, actCat, 62, TC, D0, C4 + C0, C4);
    bn(TC, mu2f, W(64), W(65), &cnts[4], N0, C4, true, actA);
    conv(2, actCat, 68, TC, D0, C4 + C0, C4);
    bn(TC, mu2f, W(69), W(70), &cnts[4], N0, C4, false, actCat);
    conv(0, actA, 63, TC, D0, C4, C4);
    bn(TC, mu2f, W(66), W(67), &cnts[4], N0, C4, false, actA);
    add_relu_bf<<<dim3(ew_blocks(N0 * C4)), dim3(256), 0, stream>>>(
        actA, actCat, actB, N0 * C4);
  }
  // --- u2_rb2 (96 -> 96): final relu(h2 + x) -> fp32 d_out ---
  conv(0, actB, 71, TC, D0, C4, C4);
  bn(TC, mu2f, W(73), W(74), &cnts[4], N0, C4, true, actA);
  conv(0, actA, 72, TC, D0, C4, C4);
  bn(TC, mu2f, W(75), W(76), &cnts[4], N0, C4, false, actA);
  add_relu_f32<<<dim3(ew_blocks(N0 * C4)), dim3(256), 0, stream>>>(
      actA, actB, (float*)d_out, N0 * C4);
}
